// Nonlocal_48722109005909
// MI455X (gfx1250) — compile-verified, matrix operand order checked by reading, not run on hardware
//
#include <hip/hip_runtime.h>
#include <hip/hip_bf16.h>

// ---------------------------------------------------------------------------
// Non-local block (self-attention) for MI455X / gfx1250.
// All matmuls via v_wmma_f32_16x16x32_bf16 (wave32 WMMA). Attention is
// flash-style (no NxPxP score materialization). GEMMs: 32x64 per-wave tiles
// (8 WMMAs / wave / K-step), double-buffered LDS, register-blocked transpose
// staging with 128-bit LDS stores, global_prefetch of upcoming tiles.
// Workspace ~262 MB.
// ---------------------------------------------------------------------------

typedef __attribute__((ext_vector_type(16))) __bf16 v16bf;
typedef __attribute__((ext_vector_type(8)))  float  v8f;

#define NB   16
#define CH   1024
#define DI   512
#define PP   1568   // 8*14*14, == 16*98 == 32*49

__device__ __forceinline__ unsigned short f32_to_bf16_u(float f) {
    unsigned int x = __float_as_uint(f);
    x += 0x7fffu + ((x >> 16) & 1u);           // round-to-nearest-even
    return (unsigned short)(x >> 16);
}

// Load one 16x32 bf16 fragment from an LDS tile stored as [16 rows][32 cols]
// (ushort, row-major). Matches the CDNA5 16-bit A-matrix layout (and the
// B-matrix layout when the tile is stored column-major, i.e. [N][K]):
//   lane L: row m = L%16, half h = L/16
//   VGPR v (0..3): K = 2v + 8h ,  VGPR v (4..7): K = 16 + 2(v-4) + 8h
__device__ __forceinline__ v16bf lds_load_frag(const unsigned short* tile, int lane) {
    const int m = lane & 15;
    const int h = (lane >> 4) << 3;            // 0 or 8
    const unsigned int* p0 = (const unsigned int*)(tile + m * 32 + h);
    const unsigned int* p1 = (const unsigned int*)(tile + m * 32 + 16 + h);
    union { unsigned int u[8]; v16bf v; } r;
#pragma unroll
    for (int i = 0; i < 4; ++i) { r.u[i] = p0[i]; r.u[4 + i] = p1[i]; }
    return r.v;
}

// ---------------------------------------------------------------------------
// fp32 -> bf16 bulk convert
// ---------------------------------------------------------------------------
__global__ void cvt_f32_bf16(const float* __restrict__ src,
                             unsigned short* __restrict__ dst, long n) {
    long i = (long)blockIdx.x * 256 + threadIdx.x;
    if (i < n) dst[i] = f32_to_bf16_u(src[i]);
}

// ---------------------------------------------------------------------------
// Generic bf16 GEMM:  C[b] = A[M,K] * B[b][K,Nn] + bias   (A shared, B batched)
// Block: 256 thr = 8 waves as 2M x 4N; wave tile 32M x 64N; workgroup tile
// 64M x 256N; K step 32; double-buffered LDS (40 KB).
// ---------------------------------------------------------------------------
template <bool F32OUT>
__global__ __launch_bounds__(256)
void gemm_bf16_kernel(const unsigned short* __restrict__ A,
                      const unsigned short* __restrict__ B,
                      const float* __restrict__ bias,
                      void* __restrict__ Cout,
                      int M, int Nn, int K, long strideB, long strideC) {
    __shared__ __align__(16) unsigned short sA[2][4 * 512];    // 64m x 32k
    __shared__ __align__(16) unsigned short sB[2][16 * 512];   // 256n x 32k col-major

    const int tid  = threadIdx.x;
    const int lane = tid & 31;
    const int wv   = tid >> 5;
    const int wm   = wv & 1;        // 32-row group
    const int wn   = wv >> 1;       // 64-col group (4 x 16)
    const int m0   = blockIdx.x * 64;
    const int n0   = blockIdx.y * 256;
    const int b    = blockIdx.z;
    const unsigned short* Bb = B + (long)b * strideB;

    auto stageA = [&](unsigned short* dst0, int kk) {
        int i0  = tid * 8;
        int row = i0 >> 5;
        int col = i0 & 31;
        const uint4 d = *(const uint4*)(A + (long)(m0 + row) * K + kk + col);
        *(uint4*)(dst0 + (row >> 4) * 512 + (row & 15) * 32 + col) = d;
    };
    // 8k x 4n register-blocked transpose; 4x ds_store_b128 per thread
    auto stageB = [&](unsigned short* dst0, int kk) {
        const int kg = (tid >> 6) * 8;       // 0,8,16,24
        const int ng = (tid & 63) * 4;       // 0..252
        unsigned short vreg[8][4];
        const unsigned short* src0 = Bb + (long)(kk + kg) * Nn + n0 + ng;
        const bool inb = (n0 + ng + 3) < Nn;
#pragma unroll
        for (int dk = 0; dk < 8; ++dk) {
            const unsigned short* src = src0 + (long)dk * Nn;
            if (inb) {
                uint2 d = *(const uint2*)src;
                vreg[dk][0] = (unsigned short)(d.x & 0xffffu);
                vreg[dk][1] = (unsigned short)(d.x >> 16);
                vreg[dk][2] = (unsigned short)(d.y & 0xffffu);
                vreg[dk][3] = (unsigned short)(d.y >> 16);
            } else {
#pragma unroll
                for (int dn = 0; dn < 4; ++dn)
                    vreg[dk][dn] = (n0 + ng + dn < Nn) ? src[dn] : (unsigned short)0;
            }
        }
#pragma unroll
        for (int dn = 0; dn < 4; ++dn) {
            int n = ng + dn;
            unsigned short* dst = dst0 + (n >> 4) * 512 + (n & 15) * 32 + kg;
            union { unsigned short s[8]; uint4 q; } pk;
#pragma unroll
            for (int dk = 0; dk < 8; ++dk) pk.s[dk] = vreg[dk][dn];
            *(uint4*)dst = pk.q;
        }
    };

    v8f acc[2][4] = {};
    stageA(sA[0], 0);
    stageB(sB[0], 0);
    int cur = 0;
    for (int kk = 0; kk < K; kk += 32) {
        __syncthreads();
        if (kk + 32 < K) {
            if (kk + 64 < K) {   // prefetch K-tile after next (global_prefetch_b8)
                __builtin_prefetch(A + (long)(m0 + (tid >> 2)) * K + kk + 64, 0, 0);
                __builtin_prefetch(Bb + (long)(kk + 64 + (tid >> 3)) * Nn + n0 + (tid & 7) * 32, 0, 0);
            }
            stageA(sA[cur ^ 1], kk + 32);
            stageB(sB[cur ^ 1], kk + 32);
        }
        v16bf a0 = lds_load_frag(sA[cur] + (wm * 2 + 0) * 512, lane);
        v16bf a1 = lds_load_frag(sA[cur] + (wm * 2 + 1) * 512, lane);
#pragma unroll
        for (int j = 0; j < 4; ++j) {
            v16bf bf = lds_load_frag(sB[cur] + (wn * 4 + j) * 512, lane);
            acc[0][j] = __builtin_amdgcn_wmma_f32_16x16x32_bf16(
                false, a0, false, bf, (short)0, acc[0][j], false, false);
            acc[1][j] = __builtin_amdgcn_wmma_f32_16x16x32_bf16(
                false, a1, false, bf, (short)0, acc[1][j], false, false);
        }
        cur ^= 1;
    }

    const int h8 = (lane >> 4) * 8;
    const int nl = lane & 15;
#pragma unroll
    for (int i = 0; i < 2; ++i) {
#pragma unroll
        for (int j = 0; j < 4; ++j) {
            int n = n0 + (wn * 4 + j) * 16 + nl;
            if (n >= Nn) continue;
#pragma unroll
            for (int r = 0; r < 8; ++r) {
                int m = m0 + (wm * 2 + i) * 16 + h8 + r;
                float v = acc[i][j][r] + bias[m];
                if (F32OUT)
                    ((float*)Cout)[(long)b * strideC + (long)m * Nn + n] = v;
                else
                    ((unsigned short*)Cout)[(long)b * strideC + (long)m * Nn + n] = f32_to_bf16_u(v);
            }
        }
    }
}

// ---------------------------------------------------------------------------
// Flash attention: one block per (query-tile of 16, batch).
// 8 waves; each wave owns a 64-channel slice of d_inner for both the score
// reduction (partial WMMAs + ds_add_f32 reduce) and the output accumulation.
// ---------------------------------------------------------------------------
__global__ __launch_bounds__(256)
void attention_kernel(const unsigned short* __restrict__ theta,
                      const unsigned short* __restrict__ phi,
                      const unsigned short* __restrict__ gmat,
                      unsigned short* __restrict__ outA) {
    __shared__ __align__(16) unsigned short sTH[16 * 512];  // theta^T: [c/32][16 q][32 c]
    __shared__ __align__(16) unsigned short sPB[32 * 512];  // phi tiles, then g tiles
    __shared__ float          sS[2 * 256];                  // [2 ksub][16 q][16 k]
    __shared__ __align__(16) unsigned short sAT[16 * 32];   // attn (A-frag layout)
    __shared__ float sM[16], sL[16], sCorr[16];

    const int tid  = threadIdx.x;
    const int lane = tid & 31;
    const int wv   = tid >> 5;
    const int q0   = blockIdx.x * 16;
    const long nOff = (long)blockIdx.y * DI * PP;
    const unsigned short* th = theta + nOff;
    const unsigned short* ph = phi   + nOff;
    const unsigned short* gg = gmat  + nOff;

    // stage theta^T for this query tile (once; persists over the key loop)
    {
        int c0 = tid * 2;
#pragma unroll
        for (int cc = 0; cc < 2; ++cc) {
            int ci = c0 + cc;
            const unsigned short* src = th + (long)ci * PP + q0;
            unsigned short* dst = sTH + (ci >> 5) * 512 + (ci & 31);
#pragma unroll
            for (int q = 0; q < 16; ++q) dst[q * 32] = src[q];
        }
    }
    if (tid < 16) { sM[tid] = -1e30f; sL[tid] = 0.f; }
    __syncthreads();

    v16bf aTH[2];
    aTH[0] = lds_load_frag(sTH + (wv * 2 + 0) * 512, lane);
    aTH[1] = lds_load_frag(sTH + (wv * 2 + 1) * 512, lane);

    v8f acc[4] = {};
    const float scale = 0.044194173824159216f;   // 512^-0.5

    for (int kb = 0; kb < PP; kb += 32) {
        __syncthreads();
        sS[tid] = 0.f; sS[tid + 256] = 0.f;
        // stage phi block: 8c x 4k register-blocked transpose, b128 LDS stores
#pragma unroll
        for (int it = 0; it < 2; ++it) {
            int a   = tid + it * 256;        // 0..511
            int kb4 = (a >> 6) * 4;          // 0,4,...,28
            int cg  = (a & 63) * 8;          // 0..504
            unsigned short vreg[8][4];
            const unsigned short* base = ph + (long)cg * PP + kb + kb4;
            if (it == 0 && kb + 32 < PP)
                __builtin_prefetch(base + 32, 0, 0);   // next key block into L2
#pragma unroll
            for (int dc = 0; dc < 8; ++dc) {
                uint2 d = *(const uint2*)(base + (long)dc * PP);
                vreg[dc][0] = (unsigned short)(d.x & 0xffffu);
                vreg[dc][1] = (unsigned short)(d.x >> 16);
                vreg[dc][2] = (unsigned short)(d.y & 0xffffu);
                vreg[dc][3] = (unsigned short)(d.y >> 16);
            }
            int tileC = (cg >> 5) * 512;
            int cIn   = cg & 31;             // 0,8,16,24
#pragma unroll
            for (int dk = 0; dk < 4; ++dk) {
                int k = kb4 + dk;
                unsigned short* dst = sPB + (k >> 4) * (16 * 512) + tileC + (k & 15) * 32 + cIn;
                union { unsigned short s[8]; uint4 q; } pk;
#pragma unroll
                for (int dc = 0; dc < 8; ++dc) pk.s[dc] = vreg[dc][dk];
                *(uint4*)dst = pk.q;
            }
        }
        __syncthreads();
        // partial scores over this wave's 64 channels, reduce across waves
#pragma unroll
        for (int s = 0; s < 2; ++s) {
            v8f ps = {};
#pragma unroll
            for (int i = 0; i < 2; ++i) {
                v16bf bf = lds_load_frag(sPB + (s * 16 + wv * 2 + i) * 512, lane);
                ps = __builtin_amdgcn_wmma_f32_16x16x32_bf16(
                    false, aTH[i], false, bf, (short)0, ps, false, false);
            }
            int qh = (lane >> 4) * 8, k = lane & 15;
#pragma unroll
            for (int r = 0; r < 8; ++r)
                atomicAdd(&sS[s * 256 + (qh + r) * 16 + k], ps[r]);
        }
        __syncthreads();
        // online softmax update (16 lanes of wave 0, one row each)
        if (tid < 16) {
            int q = tid;
            float sc[32];
            float mx = -1e30f;
#pragma unroll
            for (int k = 0; k < 32; ++k) {
                float v = sS[(k >> 4) * 256 + q * 16 + (k & 15)] * scale;
                sc[k] = v; mx = fmaxf(mx, v);
            }
            float m_old = sM[q];
            float m_new = fmaxf(m_old, mx);
            float corr  = __expf(m_old - m_new);
            float sum   = 0.f;
#pragma unroll
            for (int k = 0; k < 32; ++k) {
                float e = __expf(sc[k] - m_new);
                sum += e;
                sAT[q * 32 + k] = f32_to_bf16_u(e);
            }
            sL[q] = sL[q] * corr + sum;
            sM[q] = m_new;
            sCorr[q] = corr;
        }
        __syncthreads();
        // stage g block (reuse sPB; phi fully consumed above); contiguous b128
        {
            int c0 = tid * 2;
#pragma unroll
            for (int cc = 0; cc < 2; ++cc) {
                int ci = c0 + cc;
                const unsigned short* src = gg + (long)ci * PP + kb;
                if (kb + 32 < PP)
                    __builtin_prefetch(src + 32, 0, 0);
                unsigned short* dst = sPB + (ci >> 4) * 512 + (ci & 15) * 32;
#pragma unroll
                for (int k4 = 0; k4 < 4; ++k4)
                    *((uint4*)dst + k4) = *((const uint4*)src + k4);
            }
        }
        __syncthreads();
        // rescale accumulators, then out += attn * g^T for this wave's slice
        {
            int qh = (lane >> 4) * 8;
            float cr[8];
#pragma unroll
            for (int r = 0; r < 8; ++r) cr[r] = sCorr[qh + r];
#pragma unroll
            for (int j = 0; j < 4; ++j)
#pragma unroll
                for (int r = 0; r < 8; ++r) acc[j][r] *= cr[r];
            v16bf aA = lds_load_frag(sAT, lane);
#pragma unroll
            for (int j = 0; j < 4; ++j) {
                v16bf bf = lds_load_frag(sPB + (wv * 4 + j) * 512, lane);
                acc[j] = __builtin_amdgcn_wmma_f32_16x16x32_bf16(
                    false, aA, false, bf, (short)0, acc[j], false, false);
            }
        }
    }
    __syncthreads();
    {   // normalize and store out_attn[c][p] (bf16)
        int qh = (lane >> 4) * 8, nl = lane & 15;
#pragma unroll
        for (int j = 0; j < 4; ++j) {
            int c = wv * 64 + j * 16 + nl;
#pragma unroll
            for (int r = 0; r < 8; ++r) {
                int q = qh + r;
                float v = acc[j][r] / sL[q];
                outA[nOff + (long)c * PP + q0 + q] = f32_to_bf16_u(v);
            }
        }
    }
}

// ---------------------------------------------------------------------------
// BatchNorm stats (one block per channel) and fused BN + residual epilogue
// ---------------------------------------------------------------------------
__global__ __launch_bounds__(256)
void bn_stats_kernel(const float* __restrict__ p,
                     float* __restrict__ meanO, float* __restrict__ rstdO) {
    const int ch = blockIdx.x;
    const int NP = NB * PP;
    float s = 0.f, sq = 0.f;
    for (int i = threadIdx.x; i < NP; i += 256) {
        int n = i / PP, t = i - n * PP;
        float v = p[(long)n * (CH * (long)PP) + (long)ch * PP + t];
        s += v; sq += v * v;
    }
    __shared__ float rs[256], rq[256];
    rs[threadIdx.x] = s; rq[threadIdx.x] = sq;
    __syncthreads();
    for (int off = 128; off > 0; off >>= 1) {
        if (threadIdx.x < off) {
            rs[threadIdx.x] += rs[threadIdx.x + off];
            rq[threadIdx.x] += rq[threadIdx.x + off];
        }
        __syncthreads();
    }
    if (threadIdx.x == 0) {
        float m = rs[0] / (float)NP;
        float var = rq[0] / (float)NP - m * m;
        meanO[ch] = m;
        rstdO[ch] = rsqrtf(var + 1e-5f);
    }
}

__global__ void bn_apply_kernel(const float* __restrict__ x, const float* __restrict__ p,
                                const float* __restrict__ mean, const float* __restrict__ rstd,
                                const float* __restrict__ gamma, const float* __restrict__ beta,
                                float* __restrict__ y, long n) {
    long i = (long)blockIdx.x * 256 + threadIdx.x;
    if (i >= n) return;
    int ch = (int)((i / PP) % CH);
    y[i] = x[i] + (p[i] - mean[ch]) * rstd[ch] * gamma[ch] + beta[ch];
}

// ---------------------------------------------------------------------------
extern "C" void kernel_launch(void* const* d_in, const int* in_sizes, int n_in,
                              void* d_out, int out_size, void* d_ws, size_t ws_size,
                              hipStream_t stream) {
    (void)in_sizes; (void)n_in; (void)out_size; (void)ws_size;
    const float* x       = (const float*)d_in[0];
    const float* w_theta = (const float*)d_in[1];
    const float* b_theta = (const float*)d_in[2];
    const float* w_phi   = (const float*)d_in[3];
    const float* b_phi   = (const float*)d_in[4];
    const float* w_g     = (const float*)d_in[5];
    const float* b_g     = (const float*)d_in[6];
    const float* w_out   = (const float*)d_in[7];
    const float* b_out   = (const float*)d_in[8];
    const float* gamma   = (const float*)d_in[9];
    const float* beta    = (const float*)d_in[10];
    float* y = (float*)d_out;

    char* ws = (char*)d_ws;
    size_t off = 0;
    auto carve = [&](size_t bytes) {
        char* p = ws + off;
        off = (off + bytes + 255) & ~(size_t)255;
        return p;
    };
    unsigned short* xbf  = (unsigned short*)carve((size_t)NB * CH * PP * 2);
    unsigned short* wthb = (unsigned short*)carve((size_t)DI * CH * 2);
    unsigned short* wphb = (unsigned short*)carve((size_t)DI * CH * 2);
    unsigned short* wgb  = (unsigned short*)carve((size_t)DI * CH * 2);
    unsigned short* wob  = (unsigned short*)carve((size_t)CH * DI * 2);
    unsigned short* thb  = (unsigned short*)carve((size_t)NB * DI * PP * 2);
    unsigned short* phb  = (unsigned short*)carve((size_t)NB * DI * PP * 2);
    unsigned short* gbb  = (unsigned short*)carve((size_t)NB * DI * PP * 2);
    unsigned short* oab  = (unsigned short*)carve((size_t)NB * DI * PP * 2);
    float* pbuf  = (float*)carve((size_t)NB * CH * PP * 4);
    float* meanb = (float*)carve((size_t)CH * 4);
    float* rstdb = (float*)carve((size_t)CH * 4);

    const long nx = (long)NB * CH * PP;        // 25,690,112
    const long nw = (long)DI * CH;             // 524,288

    cvt_f32_bf16<<<dim3((unsigned)((nx + 255) / 256)), 256, 0, stream>>>(x, xbf, nx);
    cvt_f32_bf16<<<dim3((unsigned)((nw + 255) / 256)), 256, 0, stream>>>(w_theta, wthb, nw);
    cvt_f32_bf16<<<dim3((unsigned)((nw + 255) / 256)), 256, 0, stream>>>(w_phi,   wphb, nw);
    cvt_f32_bf16<<<dim3((unsigned)((nw + 255) / 256)), 256, 0, stream>>>(w_g,     wgb,  nw);
    cvt_f32_bf16<<<dim3((unsigned)((nw + 255) / 256)), 256, 0, stream>>>(w_out,   wob,  nw);

    // projections: [512x1024] x [1024x1568] per batch, bf16 out
    dim3 g1(DI / 64, (PP + 255) / 256, NB);
    gemm_bf16_kernel<false><<<g1, 256, 0, stream>>>(
        wthb, xbf, b_theta, thb, DI, PP, CH, (long)CH * PP, (long)DI * PP);
    gemm_bf16_kernel<false><<<g1, 256, 0, stream>>>(
        wphb, xbf, b_phi,   phb, DI, PP, CH, (long)CH * PP, (long)DI * PP);
    gemm_bf16_kernel<false><<<g1, 256, 0, stream>>>(
        wgb,  xbf, b_g,     gbb, DI, PP, CH, (long)CH * PP, (long)DI * PP);

    // flash attention: 98 query tiles x 16 batches
    attention_kernel<<<dim3(PP / 16, NB), 256, 0, stream>>>(thb, phb, gbb, oab);

    // output projection: [1024x512] x [512x1568] per batch, fp32 out
    dim3 g2(CH / 64, (PP + 255) / 256, NB);
    gemm_bf16_kernel<true><<<g2, 256, 0, stream>>>(
        wob, oab, b_out, pbuf, CH, PP, DI, (long)DI * PP, (long)CH * PP);

    // BN stats + fused BN/residual
    bn_stats_kernel<<<dim3(CH), 256, 0, stream>>>(pbuf, meanb, rstdb);
    bn_apply_kernel<<<dim3((unsigned)((nx + 255) / 256)), 256, 0, stream>>>(
        x, pbuf, meanb, rstdb, gamma, beta, y, nx);
}
